// LSTM_Net_24395414241326
// MI455X (gfx1250) — compile-verified, matrix operand order checked by reading, not run on hardware
//
#include <hip/hip_runtime.h>
#include <hip/hip_bf16.h>

// ---------------------------------------------------------------------------
// Fused 2-layer LSTM (H=64, Din=64, T=128, B=4096) + last-step FC(64->3).
// gfx1250 (CDNA5, wave32): gates via v_wmma_f32_16x16x32_bf16, f32 cell state
// in C-fragment registers, weights pre-packed into WMMA B-operand lane layout
// and held in LDS. Bias folded into the exp2 FMA of the activations.
// ---------------------------------------------------------------------------

typedef __attribute__((ext_vector_type(16))) __bf16        v16bf;
typedef __attribute__((ext_vector_type(8)))  float         v8f;
typedef __attribute__((ext_vector_type(4)))  unsigned int  uvec4;

#define T_LEN 128
#define NW    2            // waves per block (each wave owns 16 batch rows)

#define L2E   1.4426950408889634f
#define L2E2  2.8853900817779268f

// sigmoid(g + b) with nb = -L2E*b precomputed:  rcp(1 + exp2(fma(-L2E,g,nb)))
__device__ __forceinline__ float sig_pre(float g, float nb) {
    float e = __builtin_amdgcn_exp2f(fmaf(-L2E, g, nb));
    return __builtin_amdgcn_rcpf(1.0f + e);
}
// tanh(g + b) with nb = -2*L2E*b precomputed: 2*sigmoid(2(g+b))-1, saturates, no NaN
__device__ __forceinline__ float tanh_pre(float g, float nb) {
    float e = __builtin_amdgcn_exp2f(fmaf(-L2E2, g, nb));
    return fmaf(2.0f, __builtin_amdgcn_rcpf(1.0f + e), -1.0f);
}
__device__ __forceinline__ float tanh_fast(float c) {
    float e = __builtin_amdgcn_exp2f(-L2E2 * c);
    return fmaf(2.0f, __builtin_amdgcn_rcpf(1.0f + e), -1.0f);
}

// B-matrix fragment from LDS: pre-packed so each lane reads its own 8 dwords.
__device__ __forceinline__ v16bf ld_bfrag(const unsigned int* w, int nt, int kt, int lane) {
    const unsigned int* p = w + (((nt << 1) + kt) * 32 + lane) * 8;
    union { v16bf v; uvec4 u[2]; } r;
    r.u[0] = *(const uvec4*)(p);
    r.u[1] = *(const uvec4*)(p + 4);
    return r.v;
}

// A-matrix fragment (16x32 bf16) from a row-major [16][64] bf16 LDS tile.
// ISA layout: lane L -> row M=L%16; half=L/16: V0..3 = K[half*8 .. +7],
// V4..7 = K[16+half*8 .. +7]  (kt adds 32 to K).
__device__ __forceinline__ v16bf ld_hfrag(const __bf16* hb, int m, int half, int kt) {
    const __bf16* p = hb + m * 64 + kt * 32 + half * 8;
    union { v16bf v; uvec4 u[2]; } r;
    r.u[0] = *(const uvec4*)(p);
    r.u[1] = *(const uvec4*)(p + 16);
    return r.v;
}

// One 16x16 gate tile: zero accumulator (loop-invariant) + 4 chained WMMAs.
__device__ __forceinline__ v8f gate_tile(const unsigned int* wih, const unsigned int* whh,
                                         int nt, int lane,
                                         v16bf xa0, v16bf xa1, v16bf ha0, v16bf ha1) {
    v8f acc = {0.f, 0.f, 0.f, 0.f, 0.f, 0.f, 0.f, 0.f};
    acc = __builtin_amdgcn_wmma_f32_16x16x32_bf16(false, xa0, false, ld_bfrag(wih, nt, 0, lane), (short)0, acc, false, false);
    acc = __builtin_amdgcn_wmma_f32_16x16x32_bf16(false, xa1, false, ld_bfrag(wih, nt, 1, lane), (short)0, acc, false, false);
    acc = __builtin_amdgcn_wmma_f32_16x16x32_bf16(false, ha0, false, ld_bfrag(whh, nt, 0, lane), (short)0, acc, false, false);
    acc = __builtin_amdgcn_wmma_f32_16x16x32_bf16(false, ha1, false, ld_bfrag(whh, nt, 1, lane), (short)0, acc, false, false);
    return acc;
}

// ---------------------------------------------------------------------------
// Pack the 4 weight matrices [256,64] f32 row-major (W[g][k]) into bf16
// B-operand fragments:  B[k][n] = W[nt*16 + n][k].
// Layout per matrix: [nt(16)][kt(2)][lane(32)][dword r(8)], matrices stride 8192.
//   r<4 : k = kt*32 + half*8 + 2r       (pair k, k+1)
//   r>=4: k = kt*32 + 16 + half*8 + 2(r-4)
// ---------------------------------------------------------------------------
__global__ void pack_weights_k(const float* __restrict__ Wih0, const float* __restrict__ Whh0,
                               const float* __restrict__ Wih1, const float* __restrict__ Whh1,
                               unsigned int* __restrict__ wpack) {
    int idx = blockIdx.x * blockDim.x + threadIdx.x;
    if (idx >= 32768) return;
    int r    = idx & 7;
    int lane = (idx >> 3) & 31;
    int kt   = (idx >> 8) & 1;
    int nt   = (idx >> 9) & 15;
    int m    = idx >> 13;
    const float* W = (m == 0) ? Wih0 : (m == 1) ? Whh0 : (m == 2) ? Wih1 : Whh1;
    int n    = nt * 16 + (lane & 15);
    int half = lane >> 4;
    int klo  = kt * 32 + ((r < 4) ? (half * 8 + 2 * r) : (16 + half * 8 + 2 * (r - 4)));
    union { __bf16 h[2]; unsigned int u; } pk;
    pk.h[0] = (__bf16)W[n * 64 + klo];
    pk.h[1] = (__bf16)W[n * 64 + klo + 1];
    wpack[idx] = pk.u;
}

// ---------------------------------------------------------------------------
// Main fused kernel: 128 blocks x 64 threads (2 waves); wave w owns batch
// rows [ (blk*2+w)*16 , +16 ). Both LSTM layers stepped together over t.
// LDS/block = 141312 B -> 2 blocks per WGP (4 waves/WGP for co-execution).
// ---------------------------------------------------------------------------
__global__ void __launch_bounds__(64, 1)
lstm2_fused(const float* __restrict__ x,              // [4096,128,64]
            const unsigned int* __restrict__ wpack,   // packed weights (32768 dwords)
            const float* __restrict__ bih0, const float* __restrict__ bhh0,
            const float* __restrict__ bih1, const float* __restrict__ bhh1,
            const float* __restrict__ Wfc,  const float* __restrict__ bfc,
            float* __restrict__ out)                  // [4096,3]
{
    extern __shared__ __align__(16) unsigned char smem[];
    unsigned int* wlds = (unsigned int*)smem;                  // 131072 B
    float*        nbia = (float*)(smem + 131072);              //   2048 B (pre-scaled -bias)
    __bf16*       hbuf = (__bf16*)(smem + 133120);             //   8192 B (NW*2*16*64)

    const int tid  = threadIdx.x;
    const int wave = tid >> 5;
    const int lane = tid & 31;
    const int lm   = lane & 15;
    const int half = lane >> 4;

    for (int i = tid; i < 32768; i += 64) wlds[i] = wpack[i];
    // pre-scaled negated biases: i,f,o gates -> -L2E*(bi+bh); g gate -> -2*L2E*(bi+bh)
    for (int i = tid; i < 512; i += 64) {
        int col = i & 255;
        float s = (i < 256) ? (bih0[col] + bhh0[col]) : (bih1[col] + bhh1[col]);
        float scale = ((col >> 6) == 2) ? -L2E2 : -L2E;
        nbia[i] = scale * s;
    }
    for (int i = tid; i < NW * 2048; i += 64) hbuf[i] = (__bf16)0.0f;
    __syncthreads();

    const unsigned int* wih0 = wlds;
    const unsigned int* whh0 = wlds + 8192;
    const unsigned int* wih1 = wlds + 16384;
    const unsigned int* whh1 = wlds + 24576;

    __bf16* h0l = hbuf + wave * 2048;   // layer-1 hidden tile [16][64] bf16
    __bf16* h1l = h0l + 1024;           // layer-2 hidden tile

    const int rowbase = (blockIdx.x * NW + wave) * 16;
    const float* xrow = x + (size_t)(rowbase + lm) * T_LEN * 64;

    v8f vz = {0.f, 0.f, 0.f, 0.f, 0.f, 0.f, 0.f, 0.f};
    v8f c0[4], c1[4], h1f[4];
#pragma unroll
    for (int j = 0; j < 4; j++) { c0[j] = vz; c1[j] = vz; h1f[j] = vz; }

    for (int t = 0; t < T_LEN; ++t) {
        // ---- layer-1 input fragments straight from global x (f32 -> bf16)
        const float* xp = xrow + t * 64;
        v16bf xa0, xa1;
        {
            union { v16bf v; __bf16 h[16]; } u0, u1;
#pragma unroll
            for (int i = 0; i < 8; i++) {
                u0.h[i]     = (__bf16)xp[half * 8 + i];
                u0.h[8 + i] = (__bf16)xp[16 + half * 8 + i];
                u1.h[i]     = (__bf16)xp[32 + half * 8 + i];
                u1.h[8 + i] = (__bf16)xp[48 + half * 8 + i];
            }
            xa0 = u0.v; xa1 = u1.v;
        }
        v16bf ha0 = ld_hfrag(h0l, lm, half, 0);
        v16bf ha1 = ld_hfrag(h0l, lm, half, 1);

        // ---- layer 1 gates + cell update (i,f,g,o = N-tiles j,4+j,8+j,12+j)
        v8f hn[4];
#pragma unroll
        for (int j = 0; j < 4; j++) {
            v8f gi = gate_tile(wih0, whh0, j,      lane, xa0, xa1, ha0, ha1);
            v8f gf = gate_tile(wih0, whh0, 4 + j,  lane, xa0, xa1, ha0, ha1);
            v8f gg = gate_tile(wih0, whh0, 8 + j,  lane, xa0, xa1, ha0, ha1);
            v8f go = gate_tile(wih0, whh0, 12 + j, lane, xa0, xa1, ha0, ha1);
            float nbi = nbia[       j * 16 + lm];
            float nbf = nbia[ 64 +  j * 16 + lm];
            float nbg = nbia[128 +  j * 16 + lm];
            float nbo = nbia[192 +  j * 16 + lm];
#pragma unroll
            for (int r = 0; r < 8; r++) {
                float iv = sig_pre (gi[r], nbi);
                float fv = sig_pre (gf[r], nbf);
                float gv = tanh_pre(gg[r], nbg);
                float ov = sig_pre (go[r], nbo);
                float c  = fmaf(fv, c0[j][r], iv * gv);
                c0[j][r] = c;
                hn[j][r] = ov * tanh_fast(c);
            }
        }
        // commit new h0 (C-fragment -> row-major bf16 LDS tile)
#pragma unroll
        for (int j = 0; j < 4; j++)
#pragma unroll
            for (int r = 0; r < 8; r++)
                h0l[(r + 8 * half) * 64 + j * 16 + lm] = (__bf16)hn[j][r];
        asm volatile("s_wait_dscnt 0" ::: "memory");   // DS in-order per wave; pin compiler too

        // ---- layer 2 (input = fresh h0, hidden = previous h1)
        v16bf ia0 = ld_hfrag(h0l, lm, half, 0);
        v16bf ia1 = ld_hfrag(h0l, lm, half, 1);
        v16bf hb0 = ld_hfrag(h1l, lm, half, 0);
        v16bf hb1 = ld_hfrag(h1l, lm, half, 1);
#pragma unroll
        for (int j = 0; j < 4; j++) {
            v8f gi = gate_tile(wih1, whh1, j,      lane, ia0, ia1, hb0, hb1);
            v8f gf = gate_tile(wih1, whh1, 4 + j,  lane, ia0, ia1, hb0, hb1);
            v8f gg = gate_tile(wih1, whh1, 8 + j,  lane, ia0, ia1, hb0, hb1);
            v8f go = gate_tile(wih1, whh1, 12 + j, lane, ia0, ia1, hb0, hb1);
            float nbi = nbia[256 +        j * 16 + lm];
            float nbf = nbia[256 +  64 +  j * 16 + lm];
            float nbg = nbia[256 + 128 +  j * 16 + lm];
            float nbo = nbia[256 + 192 +  j * 16 + lm];
#pragma unroll
            for (int r = 0; r < 8; r++) {
                float iv = sig_pre (gi[r], nbi);
                float fv = sig_pre (gf[r], nbf);
                float gv = tanh_pre(gg[r], nbg);
                float ov = sig_pre (go[r], nbo);
                float c  = fmaf(fv, c1[j][r], iv * gv);
                c1[j][r] = c;
                h1f[j][r] = ov * tanh_fast(c);
            }
        }
#pragma unroll
        for (int j = 0; j < 4; j++)
#pragma unroll
            for (int r = 0; r < 8; r++)
                h1l[(r + 8 * half) * 64 + j * 16 + lm] = (__bf16)h1f[j][r];
        asm volatile("s_wait_dscnt 0" ::: "memory");
    }

    // ---- final FC: out = h1_last @ Wfc^T + bfc (f32 path, reuses weight LDS)
    __syncthreads();                       // weights dead for the whole block now
    float* hfbuf = (float*)smem;           // alias onto wlds region
#pragma unroll
    for (int j = 0; j < 4; j++)
#pragma unroll
        for (int r = 0; r < 8; r++)
            hfbuf[wave * 1024 + (r + 8 * half) * 64 + j * 16 + lm] = h1f[j][r];
    asm volatile("s_wait_dscnt 0" ::: "memory");

    if (half == 0) {
        const float* hr = hfbuf + wave * 1024 + lm * 64;
        int row = rowbase + lm;
#pragma unroll
        for (int o = 0; o < 3; o++) {
            float s = bfc[o];
            for (int k = 0; k < 64; k++) s += hr[k] * Wfc[o * 64 + k];
            out[row * 3 + o] = s;
        }
    }
}

// ---------------------------------------------------------------------------
extern "C" void kernel_launch(void* const* d_in, const int* in_sizes, int n_in,
                              void* d_out, int out_size, void* d_ws, size_t ws_size,
                              hipStream_t stream) {
    const float* x    = (const float*)d_in[0];
    const float* Wih0 = (const float*)d_in[1];
    const float* Whh0 = (const float*)d_in[2];
    const float* bih0 = (const float*)d_in[3];
    const float* bhh0 = (const float*)d_in[4];
    const float* Wih1 = (const float*)d_in[5];
    const float* Whh1 = (const float*)d_in[6];
    const float* bih1 = (const float*)d_in[7];
    const float* bhh1 = (const float*)d_in[8];
    const float* Wfc  = (const float*)d_in[9];
    const float* bfc  = (const float*)d_in[10];

    unsigned int* wpack = (unsigned int*)d_ws;   // 131072 B of scratch

    pack_weights_k<<<128, 256, 0, stream>>>(Wih0, Whh0, Wih1, Whh1, wpack);

    // dynamic LDS: 128KB packed weights + 2KB bias + 8KB h tiles = 141312 B
    // -> 2 blocks per WGP (282624 B < 320KB), 4 waves/WGP.
    size_t shmem = 131072 + 2048 + 8192;
    lstm2_fused<<<128, 64, shmem, stream>>>(x, wpack, bih0, bhh0, bih1, bhh1,
                                            Wfc, bfc, (float*)d_out);
}
